// BeamformingTransformer_84997402788007
// MI455X (gfx1250) — matrix operand order repaired and checked by reading, not compiled, not run on hardware
//
#include <hip/hip_runtime.h>
#include <hip/hip_bf16.h>

// ---------------------------------------------------------------------------
// CDNA5 (gfx1250) WMMA types & helpers
// ---------------------------------------------------------------------------
typedef __attribute__((ext_vector_type(16))) __bf16 bf16x16;
typedef __attribute__((ext_vector_type(8)))  float  f32x8;

union FragU { bf16x16 v; uint4 q[2]; };

__device__ __forceinline__ unsigned bfpack2(float a, float b) {
  unsigned ua = __float_as_uint(a); ua += 0x7FFFu + ((ua >> 16) & 1u);
  unsigned ub = __float_as_uint(b); ub += 0x7FFFu + ((ub >> 16) & 1u);
  return (ua >> 16) | (ub & 0xFFFF0000u);
}
__device__ __forceinline__ unsigned short bf1(float a) {
  unsigned ua = __float_as_uint(a); ua += 0x7FFFu + ((ua >> 16) & 1u);
  return (unsigned short)(ua >> 16);
}
// A fragment (16x32 bf16): per-lane two 16B chunks 32B apart (K=0..7/16..23 or 8..15/24..31)
__device__ __forceinline__ bf16x16 ldfragA(const unsigned short* p) {
  FragU u; u.q[0] = *(const uint4*)p; u.q[1] = *(const uint4*)(p + 16); return u.v;
}
// B fragment (32x16 bf16): per-lane column n, strided gather over K=0..7/16..23 or 8..15/24..31
__device__ __forceinline__ bf16x16 ldfragB(const unsigned short* p, int ld) {
  bf16x16 v;
#pragma unroll
  for (int s = 0; s < 16; ++s) v[s] = __builtin_bit_cast(__bf16, p[(s < 8 ? s : s + 8) * ld]);
  return v;
}
__device__ __forceinline__ f32x8 wmma_bf16(bf16x16 a, bf16x16 b, f32x8 c) {
  return __builtin_amdgcn_wmma_f32_16x16x32_bf16(false, a, false, b, (short)0, c,
                                                 false, false);
}
__device__ __forceinline__ float gelu_exact(float x) {
  return 0.5f * x * (1.0f + erff(x * 0.70710678118654752f));
}

#define GF_GELU 1
#define GF_BETA 2

// ---------------------------------------------------------------------------
// General batched GEMM:  C[b] = epi(A[b] @ W + bias [+ C] [+ R[b]])
// BLK 128x128x32, 8 waves, each wave 64x32 (4x2 WMMA tiles).
// Software-pipelined: double-buffered LDS stages + weight-stream prefetch,
// so global latency for chunk c+1 overlaps the WMMAs of chunk c.
// ---------------------------------------------------------------------------
__global__ __launch_bounds__(256) void wgemm(
    const float* __restrict__ A, int lda, long long sA,
    const float* __restrict__ W, int ldw,
    const float* __restrict__ bias,
    const float* __restrict__ R, long long sR,
    float* __restrict__ C, int ldc, long long sC,
    int K, int flags)
{
  constexpr int PA = 40;    // 80B pitch, 16B aligned
  constexpr int PB = 136;   // 272B pitch, 16B aligned
  __shared__ unsigned short As[2][128 * PA];
  __shared__ unsigned short Bs[2][32 * PB];

  const int bz = blockIdx.z;
  A += (long long)bz * sA;
  C += (long long)bz * sC;
  const float* Rb = R ? (R + (long long)bz * sR) : nullptr;
  const int m0 = blockIdx.y * 128;
  const int n0 = blockIdx.x * 128;
  const int tid = threadIdx.x;
  const int lane = tid & 31;
  const int wave = tid >> 5;
  const int wm = (wave & 1) * 64;
  const int wn = (wave >> 1) * 32;

  // per-thread tile coordinates (A: 128x32 = 1024 float4; B: 32x128 = 1024 float4)
  int ra[4], ca[4], rb[4], cb[4];
#pragma unroll
  for (int i = 0; i < 4; ++i) {
    int e = tid + i * 256;
    ra[i] = e >> 3; ca[i] = (e & 7) * 4;
    rb[i] = e >> 5; cb[i] = (e & 31) * 4;
  }

  const f32x8 zv = {0.f,0.f,0.f,0.f,0.f,0.f,0.f,0.f};
  f32x8 acc[4][2];
#pragma unroll
  for (int i = 0; i < 4; ++i)
#pragma unroll
    for (int j = 0; j < 2; ++j) acc[i][j] = zv;

  const int nch = K >> 5;
  float4 av[4], bv[4];

  // prologue: stage chunk 0
#pragma unroll
  for (int i = 0; i < 4; ++i)
    av[i] = *(const float4*)(A + (size_t)(m0 + ra[i]) * lda + ca[i]);
#pragma unroll
  for (int i = 0; i < 4; ++i)
    bv[i] = *(const float4*)(W + (size_t)rb[i] * ldw + (n0 + cb[i]));
#pragma unroll
  for (int i = 0; i < 4; ++i)
    *(uint2*)&As[0][ra[i] * PA + ca[i]] =
        make_uint2(bfpack2(av[i].x, av[i].y), bfpack2(av[i].z, av[i].w));
#pragma unroll
  for (int i = 0; i < 4; ++i)
    *(uint2*)&Bs[0][rb[i] * PB + cb[i]] =
        make_uint2(bfpack2(bv[i].x, bv[i].y), bfpack2(bv[i].z, bv[i].w));
  __syncthreads();

  for (int c = 0; c < nch; ++c) {
    const int cur = c & 1, nxt = cur ^ 1;
    const bool more = (c + 1) < nch;
    if (more) {                                    // issue next-chunk global loads early
      const int k1 = (c + 1) << 5;
#pragma unroll
      for (int i = 0; i < 4; ++i)
        av[i] = *(const float4*)(A + (size_t)(m0 + ra[i]) * lda + (k1 + ca[i]));
#pragma unroll
      for (int i = 0; i < 4; ++i)
        bv[i] = *(const float4*)(W + (size_t)(k1 + rb[i]) * ldw + (n0 + cb[i]));
      if ((c + 2) < nch) {                         // warm GL2 for the weight stream
        const int k2 = (c + 2) << 5;
#pragma unroll
        for (int i = 0; i < 4; ++i)
          __builtin_prefetch(W + (size_t)(k2 + rb[i]) * ldw + (n0 + cb[i]), 0, 1);
      }
    }
    // WMMA on current stage while loads are in flight
    bf16x16 bfr[2];
#pragma unroll
    for (int j = 0; j < 2; ++j)
      bfr[j] = ldfragB(&Bs[cur][((lane >> 4) * 8) * PB + wn + j * 16 + (lane & 15)], PB);
#pragma unroll
    for (int i = 0; i < 4; ++i) {
      bf16x16 af = ldfragA(&As[cur][(wm + i * 16 + (lane & 15)) * PA + ((lane >> 4) * 8)]);
#pragma unroll
      for (int j = 0; j < 2; ++j)
        acc[i][j] = wmma_bf16(af, bfr[j], acc[i][j]);
    }
    if (more) {                                    // stage next chunk into other buffer
#pragma unroll
      for (int i = 0; i < 4; ++i)
        *(uint2*)&As[nxt][ra[i] * PA + ca[i]] =
            make_uint2(bfpack2(av[i].x, av[i].y), bfpack2(av[i].z, av[i].w));
#pragma unroll
      for (int i = 0; i < 4; ++i)
        *(uint2*)&Bs[nxt][rb[i] * PB + cb[i]] =
            make_uint2(bfpack2(bv[i].x, bv[i].y), bfpack2(bv[i].z, bv[i].w));
    }
    __syncthreads();
  }

#pragma unroll
  for (int i = 0; i < 4; ++i) {
#pragma unroll
    for (int j = 0; j < 2; ++j) {
      int col = n0 + wn + j * 16 + (lane & 15);
      float bvv = bias ? bias[col] : 0.0f;
#pragma unroll
      for (int r = 0; r < 8; ++r) {
        int row = m0 + wm + i * 16 + (lane >> 4) * 8 + r;
        float v = acc[i][j][r];
        if (flags & GF_BETA) v += C[(size_t)row * ldc + col];
        v += bvv;
        if (Rb) v += Rb[(size_t)row * ldc + col];
        if (flags & GF_GELU) v = gelu_exact(v);
        C[(size_t)row * ldc + col] = v;
      }
    }
  }
}

// ---------------------------------------------------------------------------
// WMMA attention: one workgroup per (head, batch). lq = lk = 128, hd = 64.
// biasMode 0: cross_bias (NH,138,138);  biasMode 1: rel (NH,275) gathered.
// ---------------------------------------------------------------------------
__global__ __launch_bounds__(256) void wattn(
    const float* __restrict__ Q, const float* __restrict__ Kg,
    const float* __restrict__ V, const float* __restrict__ bias,
    int biasMode, const int* __restrict__ mask, float* __restrict__ O)
{
  extern __shared__ char smem_raw[];
  unsigned short* Qs = (unsigned short*)smem_raw;        // 128 x 72 bf16
  unsigned short* Kt = Qs + 128 * 72;                    //  64 x 136 bf16 (d-major)
  unsigned short* Vs = Kt + 64 * 136;                    // 128 x 72 bf16
  float*          S  = (float*)(Vs + 128 * 72);          // 128 x 132 f32
  unsigned short* P  = (unsigned short*)(S + 128 * 132); // 128 x 136 bf16

  const int h = blockIdx.x, b = blockIdx.y;
  const int tid = threadIdx.x, lane = tid & 31, wave = tid >> 5;
  const float* Qg = Q  + (size_t)b * 128 * 512 + h * 64;
  const float* Kb = Kg + (size_t)b * 128 * 512 + h * 64;
  const float* Vg = V  + (size_t)b * 128 * 512 + h * 64;

  for (int i = tid; i < 128 * 16; i += 256) {            // Q,V: row-major copies
    int r = i >> 4, c4 = (i & 15) * 4;
    float4 q4 = *(const float4*)(Qg + (size_t)r * 512 + c4);
    *(uint2*)&Qs[r * 72 + c4] = make_uint2(bfpack2(q4.x, q4.y), bfpack2(q4.z, q4.w));
    float4 v4 = *(const float4*)(Vg + (size_t)r * 512 + c4);
    *(uint2*)&Vs[r * 72 + c4] = make_uint2(bfpack2(v4.x, v4.y), bfpack2(v4.z, v4.w));
  }
  for (int i = tid; i < 128 * 64; i += 256) {            // K transposed (d-major)
    int r = i >> 6, d = i & 63;
    Kt[d * 136 + r] = bf1(Kb[(size_t)r * 512 + d]);
  }
  __syncthreads();

  const f32x8 zv = {0.f,0.f,0.f,0.f,0.f,0.f,0.f,0.f};
  const int qrow = wave * 16;                            // wave handles 16 q rows

  f32x8 acc[8];
#pragma unroll
  for (int j = 0; j < 8; ++j) acc[j] = zv;
#pragma unroll
  for (int c = 0; c < 2; ++c) {                          // contraction over hd=64
    bf16x16 af = ldfragA(&Qs[(qrow + (lane & 15)) * 72 + c * 32 + (lane >> 4) * 8]);
#pragma unroll
    for (int j = 0; j < 8; ++j) {
      bf16x16 kf = ldfragB(&Kt[(c * 32 + (lane >> 4) * 8) * 136 + j * 16 + (lane & 15)], 136);
      acc[j] = wmma_bf16(af, kf, acc[j]);
    }
  }
  const float* bh = bias + (biasMode == 0 ? (size_t)h * 138 * 138 : (size_t)h * 275);
  const int* mb = mask + (size_t)b * 128 * 128;
#pragma unroll
  for (int j = 0; j < 8; ++j) {
#pragma unroll
    for (int r = 0; r < 8; ++r) {
      int row = qrow + (lane >> 4) * 8 + r;
      int col = j * 16 + (lane & 15);
      float v = acc[j][r] * 0.125f;                      // 1/sqrt(64)
      if (biasMode == 0) v += bh[row * 138 + col];
      else { int ix = col - row + 137; ix = ix < 0 ? 0 : (ix > 274 ? 274 : ix); v += bh[ix]; }
      if (mb[row * 128 + col] == 0) v = -1e9f;
      S[row * 132 + col] = v;
    }
  }
  __syncthreads();
  if (tid < 128) {                                       // row softmax
    float mx = -3.0e38f;
    for (int c = 0; c < 128; ++c) mx = fmaxf(mx, S[tid * 132 + c]);
    float sum = 0.0f;
    for (int c = 0; c < 128; ++c) { float e = __expf(S[tid * 132 + c] - mx); S[tid * 132 + c] = e; sum += e; }
    float inv = 1.0f / sum;
    for (int c = 0; c < 128; ++c) P[tid * 136 + c] = bf1(S[tid * 132 + c] * inv);
  }
  __syncthreads();

  f32x8 oacc[4];
#pragma unroll
  for (int j = 0; j < 4; ++j) oacc[j] = zv;
#pragma unroll
  for (int c = 0; c < 4; ++c) {                          // contraction over 128 keys
    bf16x16 af = ldfragA(&P[(qrow + (lane & 15)) * 136 + c * 32 + (lane >> 4) * 8]);
#pragma unroll
    for (int j = 0; j < 4; ++j) {
      bf16x16 vf = ldfragB(&Vs[(c * 32 + (lane >> 4) * 8) * 72 + j * 16 + (lane & 15)], 72);
      oacc[j] = wmma_bf16(af, vf, oacc[j]);
    }
  }
  float* Og = O + (size_t)b * 128 * 512 + h * 64;
#pragma unroll
  for (int j = 0; j < 4; ++j)
#pragma unroll
    for (int r = 0; r < 8; ++r) {
      int row = qrow + (lane >> 4) * 8 + r;
      int col = j * 16 + (lane & 15);
      Og[(size_t)row * 512 + col] = oacc[j][r];
    }
}

// ---------------------------------------------------------------------------
// Small kernels
// ---------------------------------------------------------------------------
__global__ void cplx_prod(const float* __restrict__ Hr, const float* __restrict__ Hi,
                          const float* __restrict__ Wr, const float* __restrict__ Wi,
                          float* __restrict__ Pre, float* __restrict__ Pim)
{
  size_t idx = (size_t)blockIdx.x * 256 + threadIdx.x;   // b*4096 + k*64 + j
  int j = (int)(idx & 63);
  size_t bk = idx >> 6;
  size_t b = bk >> 6;
  const float* hr = Hr + bk * 64;
  const float* hi = Hi + bk * 64;
  const float* wr = Wr + (b * 64 + j) * 64;
  const float* wi = Wi + (b * 64 + j) * 64;
  float re = 0.f, im = 0.f;
  for (int n = 0; n < 64; ++n) {
    re += hr[n] * wr[n] - hi[n] * wi[n];
    im += hr[n] * wi[n] + hi[n] * wr[n];
  }
  Pre[idx] = re; Pim[idx] = im;
}

__global__ void build_tokens(const float* __restrict__ Hr, const float* __restrict__ Hi,
                             const float* __restrict__ Wr, const float* __restrict__ Wi,
                             const float* __restrict__ Pre, const float* __restrict__ Pim,
                             float* Hant, float* Want, float* Pant,
                             float* Husr, float* Wusr, float* Pusr)
{
  size_t idx = (size_t)blockIdx.x * 256 + threadIdx.x;   // b*8192 + t*64 + c
  int c = (int)(idx & 63);
  int t = (int)((idx >> 6) & 127);
  size_t b = idx >> 13;
  int tm = t & 63;
  size_t base = b * 64 * 64;
  const float* hs = (t < 64) ? Hr : Hi;
  const float* ws = (t < 64) ? Wr : Wi;
  const float* ps = (t < 64) ? Pre : Pim;
  Hant[idx] = hs[base + (size_t)c * 64 + tm];            // transpose(0,2,1) + concat
  Want[idx] = ws[base + (size_t)c * 64 + tm];
  Pant[idx] = ps[base + (size_t)c * 64 + tm];
  Husr[idx] = hs[base + (size_t)tm * 64 + c];            // plain concat
  Wusr[idx] = ws[base + (size_t)tm * 64 + c];
  Pusr[idx] = ps[base + (size_t)tm * 64 + c];
}

__global__ __launch_bounds__(256) void ln_kernel(const float* __restrict__ X,
    const float* __restrict__ g, const float* __restrict__ bta,
    float* __restrict__ Y, int width)
{
  __shared__ float red[256];
  size_t row = blockIdx.x;
  const float* x = X + row * width;
  float* y = Y + row * width;
  int tid = threadIdx.x;
  float s = 0.f;
  for (int c = tid; c < width; c += 256) s += x[c];
  red[tid] = s; __syncthreads();
  for (int st = 128; st > 0; st >>= 1) { if (tid < st) red[tid] += red[tid + st]; __syncthreads(); }
  float mean = red[0] / width;
  __syncthreads();
  float v = 0.f;
  for (int c = tid; c < width; c += 256) { float d = x[c] - mean; v += d * d; }
  red[tid] = v; __syncthreads();
  for (int st = 128; st > 0; st >>= 1) { if (tid < st) red[tid] += red[tid + st]; __syncthreads(); }
  float rstd = rsqrtf(red[0] / width + 1e-5f);
  for (int c = tid; c < width; c += 256) y[c] = (x[c] - mean) * rstd * g[c] + bta[c];
}

__global__ void gelu_kernel(float* __restrict__ X, long long n)
{
  long long i = (long long)blockIdx.x * 256 + threadIdx.x;
  if (i < n) X[i] = gelu_exact(X[i]);
}

__global__ __launch_bounds__(256) void norm_kernel(const float* __restrict__ Wv,
                                                   float* __restrict__ out)
{
  __shared__ float red[256];
  size_t row = blockIdx.x;
  const float* x = Wv + row * 8192;
  float* y = out + row * 8192;
  int tid = threadIdx.x;
  float s = 0.f;
  for (int c = tid; c < 8192; c += 256) { float v = x[c]; s += v * v; }
  red[tid] = s; __syncthreads();
  for (int st = 128; st > 0; st >>= 1) { if (tid < st) red[tid] += red[tid + st]; __syncthreads(); }
  float inv = rsqrtf(red[0]);
  for (int c = tid; c < 8192; c += 256) y[c] = x[c] * inv;
}

// ---------------------------------------------------------------------------
// Host orchestration
// Param leaf order (jax pytree, sorted dict keys) after the 6 array inputs:
//  6,7 ant_bm(b,w)  8,9 ant_ch(b,w)  10..113 ant_layers[4] x 26 leaves
//  114,115 ant_pr   116..125 out{l1.b,l1.w,l2.b,l2.w,l3.b,l3.w,ln1.b,ln1.g,ln2.b,ln2.g}
//  126,127 usr_bm   128,129 usr_ch   130..233 usr_layers  234,235 usr_pr
// Block leaf offsets: 0 cross_bias, 1/2 k.b/w, 3/4 ln1_k.b/g, 5/6 ln1_q.b/g,
//  7/8 ln1_v.b/g, 9/10 ln2.b/g, 11/12 ln_self.b/g, 13/14 m1.b/w, 15/16 m2.b/w,
//  17/18 m3.b/w, 19/20 proj.b/w, 21/22 q.b/w, 23 rel, 24/25 v.b/w
// ---------------------------------------------------------------------------
extern "C" void kernel_launch(void* const* d_in, const int* in_sizes, int n_in,
                              void* d_out, int out_size, void* d_ws, size_t ws_size,
                              hipStream_t stream)
{
  (void)in_sizes; (void)n_in; (void)out_size; (void)ws_size;
  auto f = [&](int i) { return (const float*)d_in[i]; };
  const float* Hr = f(0); const float* Hi = f(1);
  const float* Wr = f(2); const float* Wi = f(3);
  const int* umask = (const int*)d_in[4];
  const int* amask = (const int*)d_in[5];

  const long long PE   = 256LL * 64 * 64;
  const long long TOKE = 256LL * 128 * 64;
  const long long TE   = 256LL * 128 * 512;
  const long long HBE  = 256LL * 128 * 2048;

  float* wsf = (float*)d_ws;
  size_t off = 0;
  auto alloc = [&](long long n) { float* p = wsf + off; off += (size_t)((n + 255) & ~255LL); return p; };

  float* Pre = alloc(PE);  float* Pim = alloc(PE);
  float* tHa = alloc(TOKE); float* tWa = alloc(TOKE); float* tPa = alloc(TOKE);
  float* tHu = alloc(TOKE); float* tWu = alloc(TOKE); float* tPu = alloc(TOKE);
  float* xa = alloc(TE);  float* wka = alloc(TE); float* pva = alloc(TE);
  float* xu = alloc(TE);  float* wku = alloc(TE); float* pvu = alloc(TE);
  float* xT = alloc(TE);
  float* t0 = alloc(TE);  float* t1 = alloc(TE);  float* t2 = alloc(TE);
  float* Qb = alloc(TE);  float* Kb = alloc(TE);  float* Vb = alloc(TE);
  float* Yb = alloc(TE);  float* o1 = alloc(TE);  float* o2 = alloc(TE);
  float* hb1 = alloc(HBE); float* hb2 = alloc(HBE);
  float* hh1 = alloc(256LL * 2048); float* hl1 = alloc(256LL * 2048);
  float* hh2 = alloc(256LL * 1024); float* hl2 = alloc(256LL * 1024);
  float* wout = alloc(256LL * 8192);

  cplx_prod<<<4096, 256, 0, stream>>>(Hr, Hi, Wr, Wi, Pre, Pim);
  build_tokens<<<8192, 256, 0, stream>>>(Hr, Hi, Wr, Wi, Pre, Pim,
                                         tHa, tWa, tPa, tHu, tWu, tPu);

  const long long S_TOK = 128LL * 64, S_ACT = 128LL * 512, S_HB = 128LL * 2048;

  auto G = [&](const float* A, int lda, long long sA, const float* W, int ldw,
               const float* bias, const float* R, long long sR,
               float* C, int ldc, long long sC, int M, int N, int K, int flags, int batch) {
    wgemm<<<dim3(N / 128, M / 128, batch), 256, 0, stream>>>(
        A, lda, sA, W, ldw, bias, R, sR, C, ldc, sC, K, flags);
  };
  auto LN = [&](const float* X, const float* g, const float* b, float* Y, int width, int rows) {
    ln_kernel<<<rows, 256, 0, stream>>>(X, g, b, Y, width);
  };
  const int ATT_SMEM = 156672;   // Qs+Kt+Vs (bf16) + S (f32) + P (bf16)
  auto ATT = [&](const float* Qp, const float* Kp, const float* Vp,
                 const float* bias, int mode, const int* mask, float* Op) {
    wattn<<<dim3(8, 256), 256, ATT_SMEM, stream>>>(Qp, Kp, Vp, bias, mode, mask, Op);
  };

  // token embeddings (M=128, K=64, N=512, batched over 256)
  G(tHa, 64, S_TOK, f(9),   512, f(8),   nullptr, 0, xa,  512, S_ACT, 128, 512, 64, 0, 256);
  G(tWa, 64, S_TOK, f(7),   512, f(6),   nullptr, 0, wka, 512, S_ACT, 128, 512, 64, 0, 256);
  G(tPa, 64, S_TOK, f(115), 512, f(114), nullptr, 0, pva, 512, S_ACT, 128, 512, 64, 0, 256);
  G(tHu, 64, S_TOK, f(129), 512, f(128), nullptr, 0, xu,  512, S_ACT, 128, 512, 64, 0, 256);
  G(tWu, 64, S_TOK, f(127), 512, f(126), nullptr, 0, wku, 512, S_ACT, 128, 512, 64, 0, 256);
  G(tPu, 64, S_TOK, f(235), 512, f(234), nullptr, 0, pvu, 512, S_ACT, 128, 512, 64, 0, 256);

  const int ROWS = 256 * 128;
  auto runBranch = [&](float* X, float* WK, float* PV, const int* mask, int base) {
    float* cur = X; float* nxt = xT;
    for (int L = 0; L < 4; ++L) {
      auto p = [&](int i) { return (const float*)d_in[base + 26 * L + i]; };
      // cross attention
      LN(cur, p(6), p(5), t0, 512, ROWS);
      LN(WK,  p(4), p(3), t1, 512, ROWS);
      LN(PV,  p(8), p(7), t2, 512, ROWS);
      G(t0, 512, S_ACT, p(22), 512, p(21), nullptr, 0, Qb, 512, S_ACT, 128, 512, 512, 0, 256);
      G(t1, 512, S_ACT, p(2),  512, p(1),  nullptr, 0, Kb, 512, S_ACT, 128, 512, 512, 0, 256);
      G(t2, 512, S_ACT, p(25), 512, p(24), nullptr, 0, Vb, 512, S_ACT, 128, 512, 512, 0, 256);
      ATT(Qb, Kb, Vb, p(0), 0, mask, Yb);
      G(Yb, 512, S_ACT, p(20), 512, p(19), cur, S_ACT, o1, 512, S_ACT, 128, 512, 512, 0, 256);
      // self attention (relative bias)
      LN(o1, p(12), p(11), t0, 512, ROWS);
      G(t0, 512, S_ACT, p(22), 512, p(21), nullptr, 0, Qb, 512, S_ACT, 128, 512, 512, 0, 256);
      G(t0, 512, S_ACT, p(2),  512, p(1),  nullptr, 0, Kb, 512, S_ACT, 128, 512, 512, 0, 256);
      G(t0, 512, S_ACT, p(25), 512, p(24), nullptr, 0, Vb, 512, S_ACT, 128, 512, 512, 0, 256);
      ATT(Qb, Kb, Vb, p(23), 1, mask, Yb);
      G(Yb, 512, S_ACT, p(20), 512, p(19), o1, S_ACT, o2, 512, S_ACT, 128, 512, 512, 0, 256);
      // MLP
      LN(o2, p(10), p(9), t0, 512, ROWS);
      G(t0,  512,  S_ACT, p(14), 2048, p(13), nullptr, 0, hb1, 2048, S_HB, 128, 2048, 512,  GF_GELU, 256);
      G(hb1, 2048, S_HB,  p(16), 2048, p(15), nullptr, 0, hb2, 2048, S_HB, 128, 2048, 2048, GF_GELU, 256);
      G(hb2, 2048, S_HB,  p(18), 512,  p(17), o2, S_ACT, nxt, 512, S_ACT, 128, 512, 2048, 0, 256);
      float* tmp = cur; cur = nxt; nxt = tmp;   // 4 layers -> ends back in X
    }
  };
  runBranch(xa, wka, pva, amask, 10);
  runBranch(xu, wku, pvu, umask, 130);

  // output head: xf = concat(xa_flat, xu_flat) realized as two accumulating GEMMs
  G(xa, 65536, 0, f(117),                    2048, nullptr, nullptr, 0, hh1, 2048, 0, 256, 2048, 65536, 0,       1);
  G(xu, 65536, 0, f(117) + 65536LL * 2048,   2048, f(116),  nullptr, 0, hh1, 2048, 0, 256, 2048, 65536, GF_BETA, 1);
  LN(hh1, f(123), f(122), hl1, 2048, 256);
  gelu_kernel<<<(256 * 2048) / 256, 256, 0, stream>>>(hl1, 256LL * 2048);
  G(hl1, 2048, 0, f(119), 1024, f(118), nullptr, 0, hh2, 1024, 0, 256, 1024, 2048, 0, 1);
  LN(hh2, f(125), f(124), hl2, 1024, 256);
  gelu_kernel<<<(256 * 1024) / 256, 256, 0, stream>>>(hl2, 256LL * 1024);
  G(hl2, 1024, 0, f(121), 8192, f(120), nullptr, 0, wout, 8192, 0, 256, 8192, 1024, 0, 1);
  norm_kernel<<<256, 256, 0, stream>>>(wout, (float*)d_out);
}
